// GraphConv_36404142801571
// MI455X (gfx1250) — matrix operand order repaired and checked by reading, not compiled, not run on hardware
//
#include <hip/hip_runtime.h>

// ---------------- problem sizes ----------------
#define B_    128
#define CIN   32
#define NN    325
#define T_    24
#define EE    3
#define COUT  64
#define K2    128          // (EE+1)*CIN fan-in of the 1x1 conv

// ---------------- tiling ----------------
#define WT    32           // w columns per workgroup (2 WMMA N-subtiles)
#define TTL   6            // t slots per workgroup
#define NWT   11           // ceil(325/32)
#define NTT   4            // 24/6
#define M1    (CIN*TTL)    // 192 phase-1 rows (c,t)
#define NMT   (M1/16)      // 12 m-tiles == #waves
#define NK1   11           // ceil(325/32) K chunks over v
#define N2    (WT*TTL)     // 192 phase-2 columns (w,t)
#define NT2   (N2/16)      // 12 phase-2 n-tiles

// LDS pitches (elements). Row starts 16B aligned; (addr/4)%64 start banks
// spread across the 16 lanes of each half (avoid ds_load_b128 conflicts).
#define XPITCH 40          // 80B rows  -> bank start (20m)%64
#define SPITCH 40
#define HPITCH 136         // 272B rows -> bank start (4n)%64

#define NTHREADS (NMT*32)  // 384 threads = 12 wave32

typedef __attribute__((ext_vector_type(16))) __bf16 bf16x16;
typedef __attribute__((ext_vector_type(8)))  float  f32x8;
typedef __attribute__((ext_vector_type(4)))  unsigned int u32x4;

union Frag16 { bf16x16 v; u32x4 q[2]; };

__device__ __forceinline__ f32x8 wmma_bf16(bf16x16 a, bf16x16 b, f32x8 c) {
    // (neg_a, A, neg_b, B, c_mod, C, reuse_a, reuse_b)
    return __builtin_amdgcn_wmma_f32_16x16x32_bf16(false, a, false, b,
                                                   (short)0, c, false, false);
}

// fp32 -> (bf16 hi, bf16 lo) split; hi+lo carries ~16 mantissa bits.
__device__ __forceinline__ void split_bf16(float x, __bf16* hi, __bf16* lo) {
    __bf16 h = (__bf16)x;
    *hi = h;
    *lo = (__bf16)(x - (float)h);
}

// ---------------- LDS byte layout ----------------
// Region A (phase 1): xhi[M1*XPITCH], xlo[...]      = 30,720 B
// Region A'(phase 2): hhi[N2*HPITCH], hlo[...]      = 104,448 B (ALIASES region A)
// Region B:           shi[EE*WT*SPITCH], slo[...]   = 15,360 B
#define XBYTES  (M1*XPITCH*2)         // 15,360 per array
#define HBYTES  (N2*HPITCH*2)         // 52,224 per array
#define UNIONB  (2*HBYTES)            // 104,448
#define SBYTES  (EE*WT*SPITCH*2)      //  7,680 per array
#define LDS_TOTAL (UNIONB + 2*SBYTES) // 119,808 B; 2 WGs still fit a 320KB WGP

__global__ __launch_bounds__(NTHREADS)
void graphconv_fused(const float* __restrict__ x,     // [B,CIN,NN,T_]
                     const float* __restrict__ S,     // [EE,NN,NN]
                     const float* __restrict__ W,     // [COUT,K2]
                     const float* __restrict__ bias,  // [COUT]
                     float* __restrict__ out)         // [B,COUT,NN,T_]
{
    __shared__ __attribute__((aligned(16))) unsigned char lds[LDS_TOTAL];

    __bf16* xhi = (__bf16*)lds;
    __bf16* xlo = (__bf16*)(lds + XBYTES);
    __bf16* hhi = (__bf16*)lds;                 // aliases x region (barrier-protected)
    __bf16* hlo = (__bf16*)(lds + HBYTES);
    __bf16* shi = (__bf16*)(lds + UNIONB);
    __bf16* slo = (__bf16*)(lds + UNIONB + SBYTES);

    const int blk   = blockIdx.x;
    const int tt    = blk % NTT;
    const int wt    = (blk / NTT) % NWT;
    const int b     = blk / (NTT * NWT);
    const int wbase = wt * WT;
    const int tbase = tt * TTL;
    const bool wfull = (wbase + WT) <= NN;      // uniform: all 32 w columns valid

    const int tid  = threadIdx.x;
    const int lane = tid & 31;
    const int wave = tid >> 5;          // 0..11, one phase-1 m-tile per wave
    const int hsel = (lane >> 4) & 1;   // lane half (WMMA 16-bit layout)
    const int lidx = lane & 15;

    const f32x8 zero8 = {0.f,0.f,0.f,0.f,0.f,0.f,0.f,0.f};
    f32x8 acc[EE * 2] = {zero8, zero8, zero8, zero8, zero8, zero8};

    // =============== Phase 1: diff_e[(c,t), w] = sum_v x * S_e ===============
    for (int vc = 0; vc < NK1; ++vc) {
        const bool vfull = (vc * 32 + 32) <= NN;   // uniform per chunk

        // -- stage x[b, :, vchunk, ttile] -> LDS bf16 hi/lo, layout [m=(c,t)][v]
        // t-pairs: float2 loads are 8B aligned (offset = 24*idx + tbase + 2*tp).
        #pragma unroll
        for (int s = 0; s < (CIN * 32 * (TTL / 2)) / NTHREADS; ++s) {
            int i  = tid + s * NTHREADS;
            int tp = i % (TTL / 2);            // 0..2  (t = 2*tp, 2*tp+1)
            int v  = (i / (TTL / 2)) & 31;
            int c  = i / ((TTL / 2) * 32);
            float2 val;
            if (vfull) {
                val = *(const float2*)(x + ((b * CIN + c) * NN + vc * 32 + v) * T_
                                         + tbase + 2 * tp);
            } else {
                int vg = vc * 32 + v;
                val = (vg < NN)
                    ? *(const float2*)(x + ((b * CIN + c) * NN + vg) * T_ + tbase + 2 * tp)
                    : make_float2(0.f, 0.f);
            }
            __bf16 h0, l0, h1, l1;
            split_bf16(val.x, &h0, &l0);
            split_bf16(val.y, &h1, &l1);
            int m = c * TTL + 2 * tp;
            xhi[m * XPITCH + v] = h0;        xlo[m * XPITCH + v] = l0;
            xhi[(m + 1) * XPITCH + v] = h1;  xlo[(m + 1) * XPITCH + v] = l1;
        }
        // -- stage S[e, vchunk, wtile] -> LDS, layout [e][w_local][v]
        // consecutive lanes take consecutive w -> coalesced 128B runs.
        #pragma unroll
        for (int s = 0; s < (EE * WT * 32) / NTHREADS; ++s) {
            int i  = tid + s * NTHREADS;
            int wl = i & 31;
            int v  = (i >> 5) & 31;
            int e  = i >> 10;
            float val;
            if (vfull && wfull) {
                val = S[(e * NN + vc * 32 + v) * NN + wbase + wl];
            } else {
                int vg = vc * 32 + v;
                int wg = wbase + wl;
                val = (vg < NN && wg < NN) ? S[(e * NN + vg) * NN + wg] : 0.f;
            }
            __bf16 h_, l_;
            split_bf16(val, &h_, &l_);
            shi[(e * WT + wl) * SPITCH + v] = h_;
            slo[(e * WT + wl) * SPITCH + v] = l_;
        }
        __syncthreads();

        // A fragment (contiguous k order, same as B): half 0 -> K{0..15},
        // half 1 -> K{16..31}; row = mtile*16 + lidx.
        const __bf16* ap  = xhi + (wave * 16 + lidx) * XPITCH + hsel * 16;
        const __bf16* alp = xlo + (wave * 16 + lidx) * XPITCH + hsel * 16;
        Frag16 ah, al;
        ah.q[0] = *(const u32x4*)(ap);         // ds_load_b128
        ah.q[1] = *(const u32x4*)(ap + 8);
        al.q[0] = *(const u32x4*)(alp);
        al.q[1] = *(const u32x4*)(alp + 8);

        #pragma unroll
        for (int e = 0; e < EE; ++e) {
            #pragma unroll
            for (int wsub = 0; wsub < 2; ++wsub) {
                // B fragment (16-bit B 32x16): lane -> col, 16 contiguous K.
                const __bf16* bp  = shi + (e * WT + wsub * 16 + lidx) * SPITCH + hsel * 16;
                const __bf16* blp = slo + (e * WT + wsub * 16 + lidx) * SPITCH + hsel * 16;
                Frag16 bh, bl;
                bh.q[0] = *(const u32x4*)(bp);
                bh.q[1] = *(const u32x4*)(bp + 8);
                bl.q[0] = *(const u32x4*)(blp);
                bl.q[1] = *(const u32x4*)(blp + 8);
                int a = e * 2 + wsub;
                // split-bf16 fp32 emulation: hi*hi + hi*lo + lo*hi
                acc[a] = wmma_bf16(ah.v, bh.v, acc[a]);
                acc[a] = wmma_bf16(ah.v, bl.v, acc[a]);
                acc[a] = wmma_bf16(al.v, bh.v, acc[a]);
            }
        }
        __syncthreads();   // also protects h-region writes below after last chunk
    }

    // =============== Build h tile in LDS: [n=(w,t)][k=128] bf16 hi/lo ========
    // diff part: k = 32 + e*32 + c
    #pragma unroll
    for (int e = 0; e < EE; ++e) {
        #pragma unroll
        for (int wsub = 0; wsub < 2; ++wsub) {
            #pragma unroll
            for (int r = 0; r < 8; ++r) {
                int m  = wave * 16 + r + 8 * hsel;  // C-layout: row = r + 8*(lane>=16)
                int c  = m / TTL;
                int tl = m % TTL;
                int n  = (wsub * 16 + lidx) * TTL + tl;
                __bf16 h_, l_;
                split_bf16(acc[e * 2 + wsub][r], &h_, &l_);
                hhi[n * HPITCH + 32 + e * 32 + c] = h_;
                hlo[n * HPITCH + 32 + e * 32 + c] = l_;
            }
        }
    }
    // identity part: k = c (h = concat([x, diff_e...])); t-paired float2 loads.
    #pragma unroll
    for (int s = 0; s < (CIN * WT * (TTL / 2)) / NTHREADS; ++s) {
        int i  = tid + s * NTHREADS;
        int tp = i % (TTL / 2);
        int wl = (i / (TTL / 2)) & 31;
        int c  = i / ((TTL / 2) * 32);
        float2 val;
        if (wfull) {
            val = *(const float2*)(x + ((b * CIN + c) * NN + wbase + wl) * T_
                                     + tbase + 2 * tp);
        } else {
            int wg = wbase + wl;
            val = (wg < NN)
                ? *(const float2*)(x + ((b * CIN + c) * NN + wg) * T_ + tbase + 2 * tp)
                : make_float2(0.f, 0.f);
        }
        __bf16 h0, l0, h1, l1;
        split_bf16(val.x, &h0, &l0);
        split_bf16(val.y, &h1, &l1);
        int n = wl * TTL + 2 * tp;
        hhi[n * HPITCH + c] = h0;        hlo[n * HPITCH + c] = l0;
        hhi[(n + 1) * HPITCH + c] = h1;  hlo[(n + 1) * HPITCH + c] = l1;
    }
    __syncthreads();

    // =============== Phase 2: out[o,(w,t)] = W[o,k] * h[k,(w,t)] + bias ======
    const int ot = wave & 3;        // one o-tile per wave -> W frags hoisted
    const int g  = wave >> 2;       // 0..2 ; nt = g + 3*j covers 12 n-tiles

    // Preload & split W fragments for this o-tile, all 4 K-chunks.
    Frag16 wh[4], wlo[4];
    #pragma unroll
    for (int kc = 0; kc < 4; ++kc) {
        const float* wp = W + (ot * 16 + lidx) * K2 + kc * 32;
        #pragma unroll
        for (int u = 0; u < 8; ++u) {
            __bf16 h_, l_;
            split_bf16(wp[hsel * 16 + u], &h_, &l_);
            wh[kc].v[u] = h_;  wlo[kc].v[u] = l_;
            split_bf16(wp[hsel * 16 + 8 + u], &h_, &l_);
            wh[kc].v[8 + u] = h_;  wlo[kc].v[8 + u] = l_;
        }
    }
    // Bias for this wave's 8 output rows.
    float bv[8];
    #pragma unroll
    for (int r = 0; r < 8; ++r) bv[r] = bias[ot * 16 + r + 8 * hsel];

    #pragma unroll
    for (int j = 0; j < 4; ++j) {
        const int nt = g + 3 * j;
        f32x8 c2 = zero8;
        #pragma unroll
        for (int kc = 0; kc < 4; ++kc) {
            const __bf16* bp  = hhi + (nt * 16 + lidx) * HPITCH + kc * 32 + hsel * 16;
            const __bf16* blp = hlo + (nt * 16 + lidx) * HPITCH + kc * 32 + hsel * 16;
            Frag16 bh, bl;
            bh.q[0] = *(const u32x4*)(bp);
            bh.q[1] = *(const u32x4*)(bp + 8);
            bl.q[0] = *(const u32x4*)(blp);
            bl.q[1] = *(const u32x4*)(blp + 8);
            c2 = wmma_bf16(wh[kc].v, bh.v, c2);
            c2 = wmma_bf16(wh[kc].v, bl.v, c2);
            c2 = wmma_bf16(wlo[kc].v, bh.v, c2);
        }
        // bias + store (guard padded w columns; uniform fast path)
        int n  = nt * 16 + lidx;
        int wl = n / TTL;
        int tl = n % TTL;
        int wg = wbase + wl;
        int tg = tbase + tl;
        float* op = out + ((b * COUT + ot * 16 + 8 * hsel) * NN + wg) * T_ + tg;
        if (wfull || wg < NN) {
            #pragma unroll
            for (int r = 0; r < 8; ++r) {
                op[r * NN * T_] = c2[r] + bv[r];
            }
        }
    }
}

extern "C" void kernel_launch(void* const* d_in, const int* in_sizes, int n_in,
                              void* d_out, int out_size, void* d_ws, size_t ws_size,
                              hipStream_t stream) {
    (void)in_sizes; (void)n_in; (void)out_size; (void)d_ws; (void)ws_size;
    const float* x    = (const float*)d_in[0];   // [128,32,325,24]
    const float* S    = (const float*)d_in[1];   // [3,325,325]
    const float* W    = (const float*)d_in[2];   // [64,128]
    const float* bias = (const float*)d_in[3];   // [64]
    float* out = (float*)d_out;                  // [128,64,325,24]

    dim3 grid(B_ * NWT * NTT);                   // 5,632 workgroups
    dim3 block(NTHREADS);                        // 384 = 12 wave32
    graphconv_fused<<<grid, block, 0, stream>>>(x, S, W, bias, out);
}
